// GCNLayer_80006650790480
// MI455X (gfx1250) — compile-verified, matrix operand order checked by reading, not run on hardware
//
#include <hip/hip_runtime.h>
#include <hip/hip_bf16.h>
#include <math.h>

// ---------------------------------------------------------------------------
// Hyperbolic GCN layer for MI455X (gfx1250, wave32, WMMA).
// Pipeline: zero(out) -> logmap+f16split -> Wt split -> bias prep
//           -> WMMA GEMM (3-term f16 compensated) -> edge atomics -> finalize
// ---------------------------------------------------------------------------

#define D        256
#define EPSF     1e-15f
#define PROJ_MX  (1.0f - 1e-5f)
#define MAXTANH  15.0f

typedef _Float16 half4_t __attribute__((ext_vector_type(4)));
typedef _Float16 half8_t __attribute__((ext_vector_type(8)));
typedef _Float16 v16h    __attribute__((ext_vector_type(16)));
typedef float    v8f     __attribute__((ext_vector_type(8)));

__device__ __forceinline__ float wave_sum(float x) {
#pragma unroll
  for (int off = 16; off > 0; off >>= 1) x += __shfl_xor(x, off, 32);
  return x;
}

__device__ __forceinline__ float dot4(const float4& a, const float4& b) {
  return a.x * b.x + a.y * b.y + a.z * b.z + a.w * b.w;
}

// Assemble a 16x32-f16 WMMA fragment half for this lane: two contiguous
// 16B chunks at K = kb and K = kb + 16 (ISA 16-bit A/B layout, wave32).
__device__ __forceinline__ v16h load_frag(const _Float16* __restrict__ p) {
  half8_t c0 = *(const half8_t*)(p);
  half8_t c1 = *(const half8_t*)(p + 16);
  return __builtin_shufflevector(c0, c1, 0, 1, 2, 3, 4, 5, 6, 7,
                                 8, 9, 10, 11, 12, 13, 14, 15);
}

// --------------------------- zero d_out ------------------------------------
__global__ void zero_f32v4(float* __restrict__ p, int n4) {
  int i = blockIdx.x * blockDim.x + threadIdx.x;
  if (i < n4) ((float4*)p)[i] = make_float4(0.f, 0.f, 0.f, 0.f);
}

// ------------------- log_map_zero + split to f16 hi/lo ---------------------
// One wave per row (8 floats / lane). Pads rows [n_rows, n_pad) with zeros.
__global__ void logmap_split(const float* __restrict__ X,
                             _Float16* __restrict__ Xhi,
                             _Float16* __restrict__ Xlo,
                             int n_rows, int n_pad) {
  int row  = (int)((blockIdx.x * blockDim.x + threadIdx.x) >> 5);
  int lane = threadIdx.x & 31;
  if (row >= n_pad) return;

  float va[8];
  if (row < n_rows) {
    const float4* xr = (const float4*)(X + (size_t)row * D);
    float4 a = xr[lane];
    float4 b = xr[lane + 32];
    float n2 = wave_sum(dot4(a, a) + dot4(b, b));
    float n  = sqrtf(n2);
    float nc = fminf(fmaxf(n, EPSF), PROJ_MX);
    float s  = atanhf(nc) / nc;
    va[0] = a.x * s; va[1] = a.y * s; va[2] = a.z * s; va[3] = a.w * s;
    va[4] = b.x * s; va[5] = b.y * s; va[6] = b.z * s; va[7] = b.w * s;
  } else {
#pragma unroll
    for (int i = 0; i < 8; ++i) va[i] = 0.0f;
  }

  half4_t h0, h1, l0, l1;
#pragma unroll
  for (int i = 0; i < 4; ++i) {
    h0[i] = (_Float16)va[i];
    l0[i] = (_Float16)(va[i] - (float)h0[i]);
    h1[i] = (_Float16)va[4 + i];
    l1[i] = (_Float16)(va[4 + i] - (float)h1[i]);
  }
  size_t base = (size_t)row * D + lane * 4;
  *(half4_t*)(Xhi + base)       = h0;
  *(half4_t*)(Xhi + base + 128) = h1;
  *(half4_t*)(Xlo + base)       = l0;
  *(half4_t*)(Xlo + base + 128) = l1;
}

// -------------------- transpose + split weights to f16 ---------------------
__global__ void weight_t_split(const float* __restrict__ W,
                               _Float16* __restrict__ WThi,
                               _Float16* __restrict__ WTlo) {
  int k = blockIdx.x;    // input dim
  int n = threadIdx.x;   // output dim
  float w = W[(size_t)k * D + n];
  _Float16 hi = (_Float16)w;
  _Float16 lo = (_Float16)(w - (float)hi);
  WThi[(size_t)n * D + k] = hi;  // [N][K] so B-frag loads are contiguous in K
  WTlo[(size_t)n * D + k] = lo;
}

// ------------------- bias: b = proj(exp_map_zero(bias)) --------------------
__global__ void bias_prep(const float* __restrict__ bias,
                          float* __restrict__ bvec,
                          float* __restrict__ nb2out) {
  int lane = threadIdx.x;  // 32 threads, 8 floats each
  const float4* bp = (const float4*)bias;
  float4 a = bp[lane];
  float4 b = bp[lane + 32];
  float n2 = wave_sum(dot4(a, a) + dot4(b, b));
  float n  = fmaxf(sqrtf(n2), EPSF);
  float t  = tanhf(fminf(n, MAXTANH));
  float s  = t / n;
  float p  = fminf(1.0f, PROJ_MX / fmaxf(t, EPSF));
  float sc = s * p;
  float4 oa = make_float4(a.x * sc, a.y * sc, a.z * sc, a.w * sc);
  float4 ob = make_float4(b.x * sc, b.y * sc, b.z * sc, b.w * sc);
  ((float4*)bvec)[lane]      = oa;
  ((float4*)bvec)[lane + 32] = ob;
  if (lane == 0) *nb2out = (t * p) * (t * p);
}

// ------------------------------- WMMA GEMM ---------------------------------
// Y[m, n] = sum_k Xlog[m, k] * W[k, n], f16 split (Ahi*Bhi + Alo*Bhi + Ahi*Blo),
// f32 accumulate. One wave handles one 16-row M-tile x four 16-col N-tiles.
__global__ void wmma_gemm(const _Float16* __restrict__ Ahi,
                          const _Float16* __restrict__ Alo,
                          const _Float16* __restrict__ Bhi,
                          const _Float16* __restrict__ Blo,
                          float* __restrict__ Y, int n_mtiles) {
  int gw    = (int)((blockIdx.x * blockDim.x + threadIdx.x) >> 5);
  int lane  = threadIdx.x & 31;
  int mtile = gw >> 2;
  int ngrp  = gw & 3;
  if (mtile >= n_mtiles) return;

  int mlane = (mtile << 4) + (lane & 15);  // A row carried by this lane
  int khalf = (lane >> 4) << 3;            // 0 or 8: K sub-chunk select
  const _Float16* arow_hi = Ahi + (size_t)mlane * D;
  const _Float16* arow_lo = Alo + (size_t)mlane * D;

  v8f acc[4] = {};

#pragma unroll
  for (int ks = 0; ks < 8; ++ks) {
    int kb = ks * 32 + khalf;
    v16h ahi = load_frag(arow_hi + kb);
    v16h alo = load_frag(arow_lo + kb);
#pragma unroll
    for (int t = 0; t < 4; ++t) {
      int col = (ngrp << 6) + (t << 4) + (lane & 15);  // B column for lane
      v16h bhi = load_frag(Bhi + (size_t)col * D + kb);
      v16h blo = load_frag(Blo + (size_t)col * D + kb);
      acc[t] = __builtin_amdgcn_wmma_f32_16x16x32_f16(
          false, ahi, false, bhi, (short)0, acc[t], false, false);
      acc[t] = __builtin_amdgcn_wmma_f32_16x16x32_f16(
          false, alo, false, bhi, (short)0, acc[t], false, false);
      acc[t] = __builtin_amdgcn_wmma_f32_16x16x32_f16(
          false, ahi, false, blo, (short)0, acc[t], false, false);
    }
  }

  // C/D layout: acc[v] in lane l is C[v + 8*(l>>4)][l&15]
  int m0 = (mtile << 4) + ((lane >> 4) << 3);
  int n0 = (ngrp << 6) + (lane & 15);
#pragma unroll
  for (int t = 0; t < 4; ++t) {
    int ncol = n0 + (t << 4);
#pragma unroll
    for (int v = 0; v < 8; ++v)
      Y[(size_t)(m0 + v) * D + ncol] = acc[t][v];
  }
}

// ------------------- edge scatter: out[row] += val * Y[col] ----------------
// 64 threads per edge, float4 gather (Y is L2-resident: 100 MB < 192 MB L2),
// 4x global_atomic_add_f32 scatter.
__global__ void spmm_atomic(const float* __restrict__ Y,
                            const int* __restrict__ erow,
                            const int* __restrict__ ecol,
                            const float* __restrict__ eval,
                            float* __restrict__ out, int n_edges) {
  int gid = blockIdx.x * blockDim.x + threadIdx.x;
  int e = gid >> 6;
  if (e >= n_edges) return;
  int j = (gid & 63) << 2;
  int r = erow[e];
  int c = ecol[e];
  float v = eval[e];
  float4 g = *(const float4*)(Y + (size_t)c * D + j);
  float* o = out + (size_t)r * D + j;
  atomicAdd(o + 0, v * g.x);
  atomicAdd(o + 1, v * g.y);
  atomicAdd(o + 2, v * g.z);
  atomicAdd(o + 3, v * g.w);
}

// ---------- finalize: proj(exp_map(agg)), Mobius(+b), proj; in place -------
__global__ void finalize_rows(float* __restrict__ out,
                              const float* __restrict__ bvec,
                              const float* __restrict__ nb2p, int n_rows) {
  int row  = (int)((blockIdx.x * blockDim.x + threadIdx.x) >> 5);
  int lane = threadIdx.x & 31;
  if (row >= n_rows) return;

  float* orow = out + (size_t)row * D;
  float4 a  = ((const float4*)orow)[lane];
  float4 b  = ((const float4*)orow)[lane + 32];
  float4 ba = ((const float4*)bvec)[lane];
  float4 bb = ((const float4*)bvec)[lane + 32];
  float nb2 = *nb2p;

  // exp_map_zero + projection (norm of result is tanh analytically)
  float n2 = wave_sum(dot4(a, a) + dot4(b, b));
  float n  = fmaxf(sqrtf(n2), EPSF);
  float t  = tanhf(fminf(n, MAXTANH));
  float s  = t / n;
  float p  = fminf(1.0f, PROJ_MX / fmaxf(t, EPSF));
  float us = s * p;                 // u = us * agg
  float nu2 = (t * p) * (t * p);    // ||u||^2

  // Mobius addition u (+) b
  float dp  = us * (dot4(a, ba) + dot4(b, bb));
  float dot = wave_sum(dp);
  float den = 1.0f + 2.0f * dot + nu2 * nb2 + EPSF;
  float w1  = (1.0f + 2.0f * dot + nb2) * us / den;
  float w2  = (1.0f - nu2) / den;

  float4 o1 = make_float4(w1 * a.x + w2 * ba.x, w1 * a.y + w2 * ba.y,
                          w1 * a.z + w2 * ba.z, w1 * a.w + w2 * ba.w);
  float4 o2 = make_float4(w1 * b.x + w2 * bb.x, w1 * b.y + w2 * bb.y,
                          w1 * b.z + w2 * bb.z, w1 * b.w + w2 * bb.w);

  // final projection back into the ball
  float on2 = wave_sum(dot4(o1, o1) + dot4(o2, o2));
  float on  = fmaxf(sqrtf(on2), EPSF);
  float q   = fminf(1.0f, PROJ_MX / on);
  o1.x *= q; o1.y *= q; o1.z *= q; o1.w *= q;
  o2.x *= q; o2.y *= q; o2.z *= q; o2.w *= q;
  ((float4*)orow)[lane]      = o1;
  ((float4*)orow)[lane + 32] = o2;
}

// ---------------------------------------------------------------------------
extern "C" void kernel_launch(void* const* d_in, const int* in_sizes, int n_in,
                              void* d_out, int out_size, void* d_ws, size_t ws_size,
                              hipStream_t stream) {
  const float* X    = (const float*)d_in[0];
  const float* W    = (const float*)d_in[1];
  const float* bias = (const float*)d_in[2];
  const int*   erow = (const int*)d_in[3];
  const int*   ecol = (const int*)d_in[4];
  const float* eval = (const float*)d_in[5];
  float* out = (float*)d_out;

  const int n_nodes  = in_sizes[0] / D;
  const int n_edges  = in_sizes[3];
  const int n_mtiles = (n_nodes + 15) / 16;
  const int n_pad    = n_mtiles * 16;

  // workspace carve-up (256B aligned)
  char* ws = (char*)d_ws;
  size_t off = 0;
  auto carve = [&](size_t bytes) -> void* {
    void* p = ws + off;
    off = (off + bytes + 255) & ~(size_t)255;
    return p;
  };
  _Float16* Xhi  = (_Float16*)carve((size_t)n_pad * D * sizeof(_Float16));
  _Float16* Xlo  = (_Float16*)carve((size_t)n_pad * D * sizeof(_Float16));
  _Float16* WThi = (_Float16*)carve((size_t)D * D * sizeof(_Float16));
  _Float16* WTlo = (_Float16*)carve((size_t)D * D * sizeof(_Float16));
  float*    Y    = (float*)carve((size_t)n_pad * D * sizeof(float));
  float*    bvec = (float*)carve((size_t)D * sizeof(float));
  float*    nb2  = (float*)carve(sizeof(float));

  // 1) zero the accumulation target (d_out)
  {
    int n4 = n_nodes * (D / 4);
    zero_f32v4<<<(n4 + 255) / 256, 256, 0, stream>>>(out, n4);
  }
  // 2) log-map rows -> f16 hi/lo (one wave per row, 8 rows per block)
  logmap_split<<<(n_pad + 7) / 8, 256, 0, stream>>>(X, Xhi, Xlo, n_nodes, n_pad);
  // 3) transpose + split weights
  weight_t_split<<<D, D, 0, stream>>>(W, WThi, WTlo);
  // 4) bias point on the ball
  bias_prep<<<1, 32, 0, stream>>>(bias, bvec, nb2);
  // 5) WMMA GEMM: 4 waves per M-tile (each owns 64 output columns)
  {
    int waves  = n_mtiles * 4;
    int blocks = (waves * 32 + 255) / 256;
    wmma_gemm<<<blocks, 256, 0, stream>>>(Xhi, Xlo, WThi, WTlo, Y, n_mtiles);
  }
  // 6) edge gather (L2-resident) + atomic scatter into d_out
  {
    long long thr = (long long)n_edges * 64;
    int blocks = (int)((thr + 255) / 256);
    spmm_atomic<<<blocks, 256, 0, stream>>>(Y, erow, ecol, eval, out, n_edges);
  }
  // 7) exp-map + projection + Mobius bias + projection, in place
  finalize_rows<<<(n_nodes + 7) / 8, 256, 0, stream>>>(out, bvec, nb2, n_nodes);
}
